// MultiHeadAttention_43267500540068
// MI455X (gfx1250) — matrix operand order repaired and checked by reading, not compiled, not run on hardware
//
#include <hip/hip_runtime.h>

// ---------------------------------------------------------------------------
// MHA forward for MI455X (gfx1250, wave32).  bf16 WMMA everywhere, fp32 accum.
// B=4, T=2048, D=1024, H=16, DH=64.  All dims compile-time.
// ---------------------------------------------------------------------------

namespace {
constexpr int Bc  = 4;
constexpr int Tc  = 2048;
constexpr int Dc  = 1024;
constexpr int Hc  = 16;
constexpr int DHc = 64;

constexpr int Mg  = Bc * Tc;  // 8192  gemm M
constexpr int Ng  = Dc;       // 1024  gemm N
constexpr int Kg  = Dc;       // 1024  gemm K

constexpr int LDT = 72;   // padded LDS row stride in bf16 elements (144 B, 16B-aligned)
constexpr int BM  = 128;  // gemm block tile M
constexpr int BN  = 128;  // gemm block tile N
constexpr int KC  = 64;   // gemm K chunk
constexpr int QB  = 128;  // attention q rows per block
constexpr int KB  = 64;   // attention keys per chunk
}

typedef __attribute__((ext_vector_type(16))) __bf16 v16bf;
typedef __attribute__((ext_vector_type(8)))  float  v8f;

__device__ __forceinline__ v8f wmma_bf16(v16bf a, v16bf b, v8f c) {
  // D = A(16x32 bf16) * B(32x16 bf16) + C(16x16 f32)
  return __builtin_amdgcn_wmma_f32_16x16x32_bf16(false, a, false, b, (short)0, c,
                                                 false, false);
}

__device__ __forceinline__ v8f zero8() {
  v8f z;
#pragma unroll
  for (int i = 0; i < 8; ++i) z[i] = 0.0f;
  return z;
}

// A-fragment (16x32, MxK): lanes 0-15 row M=lane, K in {0..7, 16..23};
// lanes 16-31 row M=lane-16, K in {8..15, 24..31}; 2 bf16 per VGPR.
__device__ __forceinline__ v16bf load_frag_a(const __bf16* base, int lane) {
  const int half = lane >> 4;
  const int m    = lane & 15;
  union { v16bf v; unsigned u[8]; } f;
  const unsigned* row = (const unsigned*)(base + m * LDT);
#pragma unroll
  for (int i = 0; i < 8; ++i) {
    const int kpair = (i < 4) ? (half * 4 + i) : (8 + half * 4 + (i - 4));
    f.u[i] = row[kpair];
  }
  return f.v;
}

// B-fragment (32x16, KxN) read from B^T storage Bt[n][k]:
// lanes 0-15 col N=lane, K in {0..7, 16..23}; lanes 16-31 col N=lane-16, K in {8..15, 24..31}.
__device__ __forceinline__ v16bf load_frag_b(const __bf16* base, int lane) {
  const int half = lane >> 4;
  const int n    = lane & 15;
  union { v16bf v; unsigned u[8]; } f;
  const unsigned* row = (const unsigned*)(base + n * LDT);
#pragma unroll
  for (int i = 0; i < 8; ++i) f.u[i] = row[(i < 4) ? (half * 4 + i) : (8 + half * 4 + (i - 4))];
  return f.v;
}

// ---------------------------------------------------------------------------
// fp32 -> bf16 conversion (4 elements / thread)
// ---------------------------------------------------------------------------
__global__ void cvt_f32_bf16(const float* __restrict__ in, __bf16* __restrict__ out) {
  const size_t i = ((size_t)blockIdx.x * blockDim.x + threadIdx.x) * 4;
  const float4 f = *(const float4*)(in + i);
  alignas(8) __bf16 o[4] = { (__bf16)f.x, (__bf16)f.y, (__bf16)f.z, (__bf16)f.w };
  *(uint2*)(out + i) = *(const uint2*)o;
}

// ---------------------------------------------------------------------------
// GEMM:  Out[m][n] = sum_k A[m][k] * Bt[n][k]     (i.e. y = x @ W^T)
// MODE 0: bf16 out, remapped to [B][H][T][DH] head-split layout
// MODE 1: f32 out, plain [Mg][Ng]
// MODE 2: like MODE 0 but scaled by 1/sqrt(DH)=0.125 (Q projection; exact in bf16)
// Block tile 128x128, 8 waves (4Mx2N), each wave 32x64:
//   per 32-wide k-step: 2 A-frags + 4 B-frags (12 ds_load_b128) -> 8 WMMAs.
// ---------------------------------------------------------------------------
template <int MODE>
__global__ __launch_bounds__(256) void gemm_bf16(const __bf16* __restrict__ A,
                                                 const __bf16* __restrict__ Bt,
                                                 void* __restrict__ Out) {
  __shared__ __bf16 Asm[BM][LDT];
  __shared__ __bf16 Bsm[BN][LDT];

  const int tid  = threadIdx.x;
  const int wave = tid >> 5;
  const int lane = tid & 31;
  const int m0   = blockIdx.y * BM;
  const int n0   = blockIdx.x * BN;
  const int wm   = (wave >> 1) * 32;   // wave M offset in block tile (0..96)
  const int wn   = (wave & 1) * 64;    // wave N offset in block tile (0,64)

  v8f acc[2][4];
#pragma unroll
  for (int i = 0; i < 2; ++i)
#pragma unroll
    for (int j = 0; j < 4; ++j) acc[i][j] = zero8();

  const int lr = tid >> 3;        // 0..31 row within pass
  const int lc = (tid & 7) * 8;   // 0..56 col (bf16 elements, 16B chunks)

  for (int kc = 0; kc < Kg; kc += KC) {
    // stage A tile: 128 rows x 64 k (16 B per thread per pass)
#pragma unroll
    for (int p = 0; p < 4; ++p) {
      const int r = p * 32 + lr;
      *(uint4*)&Asm[r][lc] = *(const uint4*)&A[(size_t)(m0 + r) * Kg + kc + lc];
      __builtin_prefetch(&A[(size_t)(m0 + r) * Kg + kc + KC + lc], 0, 1);
    }
    // stage B^T tile: 128 rows x 64 k
#pragma unroll
    for (int p = 0; p < 4; ++p) {
      const int r = p * 32 + lr;
      *(uint4*)&Bsm[r][lc] = *(const uint4*)&Bt[(size_t)(n0 + r) * Kg + kc + lc];
      __builtin_prefetch(&Bt[(size_t)(n0 + r) * Kg + kc + KC + lc], 0, 1);
    }
    __syncthreads();

#pragma unroll
    for (int kk = 0; kk < KC; kk += 32) {
      const v16bf a0 = load_frag_a(&Asm[wm][kk],      lane);
      const v16bf a1 = load_frag_a(&Asm[wm + 16][kk], lane);
      v16bf bf[4];
#pragma unroll
      for (int j = 0; j < 4; ++j) bf[j] = load_frag_b(&Bsm[wn + j * 16][kk], lane);
#pragma unroll
      for (int j = 0; j < 4; ++j) {
        acc[0][j] = wmma_bf16(a0, bf[j], acc[0][j]);
        acc[1][j] = wmma_bf16(a1, bf[j], acc[1][j]);
      }
    }
    __syncthreads();
  }

  // epilogue: C/D layout -> global (all index math is shifts/adds)
  const int nn = lane & 15;
  const int mb = (lane >> 4) * 8;
#pragma unroll
  for (int i = 0; i < 2; ++i)
#pragma unroll
    for (int j = 0; j < 4; ++j)
#pragma unroll
      for (int r = 0; r < 8; ++r) {
        const int m = m0 + wm + i * 16 + mb + r;
        const int n = n0 + wn + j * 16 + nn;
        float val = acc[i][j][r];
        if (MODE == 2) val *= 0.125f;    // fold 1/sqrt(DH) into Q (exact pow2)
        if (MODE == 0 || MODE == 2) {
          const int b = m >> 11, t = m & (Tc - 1);
          const int h = n >> 6,  d = n & 63;
          ((__bf16*)Out)[(((size_t)b * Hc + h) * Tc + t) * DHc + d] = (__bf16)val;
        } else {
          ((float*)Out)[(size_t)m * Ng + n] = val;
        }
      }
}

// ---------------------------------------------------------------------------
// Causal flash attention over head-split bf16 tensors [B][H][T][DH].
// Block = 128 q rows x one (b,h); 8 waves x 16 q rows each.
// Q is pre-scaled by 1/sqrt(DH) in its projection epilogue.
// Keys streamed in chunks of 64 through LDS; online softmax; O = P*V via WMMA.
// Writes attention output to Ao in [B][T][D] bf16 layout for the final GEMM.
// ---------------------------------------------------------------------------
__global__ __launch_bounds__(256) void attn_kernel(const __bf16* __restrict__ Qh,
                                                   const __bf16* __restrict__ Kh,
                                                   const __bf16* __restrict__ Vh,
                                                   __bf16* __restrict__ Ao) {
  __shared__ __bf16 Qsm[QB][LDT];        // 128 x 64 q tile
  __shared__ __bf16 Ksm[KB][LDT];        // 64 keys x 64 dh   (B^T for Q*K^T)
  __shared__ __bf16 Vtsm[DHc][LDT];      // V transposed: [dh][key] (B^T for P*V)
  __shared__ __bf16 Psm[8][16][LDT];     // per-wave P staging (C-layout -> A-frag)

  const int bh = blockIdx.y;             // 0..B*H-1
  const int b  = bh >> 4;
  const int h  = bh & 15;
  const int q0 = blockIdx.x * QB;

  const int tid  = threadIdx.x;
  const int wave = tid >> 5;
  const int lane = tid & 31;
  const int wm   = wave * 16;            // wave q-row offset within tile
  const int nn   = lane & 15;
  const int mb   = (lane >> 4) * 8;
  const int lr   = tid >> 3;
  const int lc   = (tid & 7) * 8;

  const __bf16* Qp = Qh + (size_t)bh * Tc * DHc;
  const __bf16* Kp = Kh + (size_t)bh * Tc * DHc;
  const __bf16* Vp = Vh + (size_t)bh * Tc * DHc;

  // stage Q tile and build per-wave Q fragments (held in registers throughout)
#pragma unroll
  for (int p = 0; p < 4; ++p) {
    const int r = p * 32 + lr;
    *(uint4*)&Qsm[r][lc] = *(const uint4*)&Qp[(size_t)(q0 + r) * DHc + lc];
  }
  __syncthreads();
  const v16bf qf0 = load_frag_a(&Qsm[wm][0],  lane);
  const v16bf qf1 = load_frag_a(&Qsm[wm][32], lane);

  v8f o[4];
#pragma unroll
  for (int jd = 0; jd < 4; ++jd) o[jd] = zero8();

  const float NEG_INF = -__builtin_inff();
  float mrow[8], lrow[8];                // per-lane stats for rows mb..mb+7
#pragma unroll
  for (int r = 0; r < 8; ++r) { mrow[r] = NEG_INF; lrow[r] = 0.0f; }

  const int rowbase = q0 + wm + mb;
  const int kend    = q0 + QB;           // causal: keys < q-block end

  for (int kc = 0; kc < kend; kc += KB) {
    __syncthreads();                     // previous chunk's LDS fully consumed
    // stage K chunk [key][dh]
#pragma unroll
    for (int p = 0; p < 2; ++p) {
      const int r = p * 32 + lr;
      *(uint4*)&Ksm[r][lc] = *(const uint4*)&Kp[(size_t)(kc + r) * DHc + lc];
    }
    // stage V chunk transposed -> Vtsm[dh][key]
#pragma unroll
    for (int p = 0; p < 2; ++p) {
      const int r = p * 32 + lr;
      const uint4 raw = *(const uint4*)&Vp[(size_t)(kc + r) * DHc + lc];
      const __bf16* e = (const __bf16*)&raw;
#pragma unroll
      for (int i = 0; i < 8; ++i) Vtsm[lc + i][r] = e[i];
    }
    __syncthreads();

    // S = Q * K^T  (16 x 64 per wave, 4 n-tiles); Q pre-scaled
    v8f s[4];
#pragma unroll
    for (int j = 0; j < 4; ++j) {
      const v16bf kb0 = load_frag_b(&Ksm[j * 16][0],  lane);
      const v16bf kb1 = load_frag_b(&Ksm[j * 16][32], lane);
      v8f c = zero8();
      c = wmma_bf16(qf0, kb0, c);
      c = wmma_bf16(qf1, kb1, c);
      s[j] = c;
    }

    // causal mask
#pragma unroll
    for (int j = 0; j < 4; ++j) {
      const int col = kc + j * 16 + nn;
#pragma unroll
      for (int r = 0; r < 8; ++r) {
        if (col > rowbase + r) s[j][r] = NEG_INF;
      }
    }

    // online softmax: row max / exp / row sum (reduce across 16-lane half)
#pragma unroll
    for (int r = 0; r < 8; ++r) {
      float mx = fmaxf(fmaxf(s[0][r], s[1][r]), fmaxf(s[2][r], s[3][r]));
      mx = fmaxf(mx, __shfl_xor(mx, 1));
      mx = fmaxf(mx, __shfl_xor(mx, 2));
      mx = fmaxf(mx, __shfl_xor(mx, 4));
      mx = fmaxf(mx, __shfl_xor(mx, 8));
      const float newm = fmaxf(mrow[r], mx);

      float rs = 0.0f;
#pragma unroll
      for (int j = 0; j < 4; ++j) {
        const float p = __expf(s[j][r] - newm);
        s[j][r] = p;
        rs += p;
      }
      rs += __shfl_xor(rs, 1);
      rs += __shfl_xor(rs, 2);
      rs += __shfl_xor(rs, 4);
      rs += __shfl_xor(rs, 8);

      const float alpha = __expf(mrow[r] - newm);
      lrow[r] = lrow[r] * alpha + rs;
      mrow[r] = newm;
#pragma unroll
      for (int jd = 0; jd < 4; ++jd) o[jd][r] *= alpha;
    }

    // P (C-layout) -> per-wave LDS tile -> A-fragments
#pragma unroll
    for (int j = 0; j < 4; ++j)
#pragma unroll
      for (int r = 0; r < 8; ++r)
        Psm[wave][mb + r][j * 16 + nn] = (__bf16)s[j][r];

    const v16bf pf0 = load_frag_a(&Psm[wave][0][0],  lane);
    const v16bf pf1 = load_frag_a(&Psm[wave][0][32], lane);

    // O += P * V   (4 d-tiles)
#pragma unroll
    for (int jd = 0; jd < 4; ++jd) {
      const v16bf vb0 = load_frag_b(&Vtsm[jd * 16][0],  lane);
      const v16bf vb1 = load_frag_b(&Vtsm[jd * 16][32], lane);
      o[jd] = wmma_bf16(pf0, vb0, o[jd]);
      o[jd] = wmma_bf16(pf1, vb1, o[jd]);
    }
  }

  // normalize and write to [B][T][D] bf16
#pragma unroll
  for (int jd = 0; jd < 4; ++jd)
#pragma unroll
    for (int r = 0; r < 8; ++r) {
      const int t   = q0 + wm + mb + r;
      const int col = h * DHc + jd * 16 + nn;
      const float val = o[jd][r] / lrow[r];
      Ao[((size_t)b * Tc + t) * Dc + col] = (__bf16)val;
    }
}

// ---------------------------------------------------------------------------
// Host-side orchestration (all launches on `stream`; workspace-only scratch)
// ---------------------------------------------------------------------------
extern "C" void kernel_launch(void* const* d_in, const int* in_sizes, int n_in,
                              void* d_out, int out_size, void* d_ws, size_t ws_size,
                              hipStream_t stream) {
  (void)in_sizes; (void)n_in; (void)out_size; (void)ws_size;

  const float* q  = (const float*)d_in[0];
  const float* k  = (const float*)d_in[1];
  const float* v  = (const float*)d_in[2];
  const float* wq = (const float*)d_in[3];
  const float* wk = (const float*)d_in[4];
  const float* wv = (const float*)d_in[5];
  const float* wo = (const float*)d_in[6];
  // d_in[7] = causal mask (implicit in attention kernel)

  const size_t ND = (size_t)Bc * Tc * Dc;   // 8,388,608 activations
  const size_t NW = (size_t)Dc * Dc;        // 1,048,576 weight elements

  char*  ws  = (char*)d_ws;
  size_t off = 0;
  auto take = [&](size_t bytes) -> char* {
    char* p = ws + off;
    off += (bytes + 255) & ~(size_t)255;
    return p;
  };

  __bf16* qb  = (__bf16*)take(ND * 2);
  __bf16* kb  = (__bf16*)take(ND * 2);
  __bf16* vb  = (__bf16*)take(ND * 2);
  __bf16* wqb = (__bf16*)take(NW * 2);
  __bf16* wkb = (__bf16*)take(NW * 2);
  __bf16* wvb = (__bf16*)take(NW * 2);
  __bf16* wob = (__bf16*)take(NW * 2);
  __bf16* Qh  = (__bf16*)take(ND * 2);      // [B][H][T][DH], pre-scaled by 0.125
  __bf16* Kh  = (__bf16*)take(ND * 2);
  __bf16* Vh  = (__bf16*)take(ND * 2);
  __bf16* Ao  = (__bf16*)take(ND * 2);      // [B][T][D]

  const int ablk = (int)(ND / 4 / 256);     // 8192
  const int wblk = (int)(NW / 4 / 256);     // 1024
  cvt_f32_bf16<<<ablk, 256, 0, stream>>>(q,  qb);
  cvt_f32_bf16<<<ablk, 256, 0, stream>>>(k,  kb);
  cvt_f32_bf16<<<ablk, 256, 0, stream>>>(v,  vb);
  cvt_f32_bf16<<<wblk, 256, 0, stream>>>(wq, wqb);
  cvt_f32_bf16<<<wblk, 256, 0, stream>>>(wk, wkb);
  cvt_f32_bf16<<<wblk, 256, 0, stream>>>(wv, wvb);
  cvt_f32_bf16<<<wblk, 256, 0, stream>>>(wo, wob);

  const dim3 gg(Ng / BN, Mg / BM);          // (8, 64)
  gemm_bf16<2><<<gg, 256, 0, stream>>>(qb, wqb, Qh);   // Q: scaled
  gemm_bf16<0><<<gg, 256, 0, stream>>>(kb, wkb, Kh);
  gemm_bf16<0><<<gg, 256, 0, stream>>>(vb, wvb, Vh);

  attn_kernel<<<dim3(Tc / QB, Bc * Hc), 256, 0, stream>>>(Qh, Kh, Vh, Ao);

  gemm_bf16<1><<<gg, 256, 0, stream>>>(Ao, wob, d_out);
}
